// TurboQuantAttention_59639915872720
// MI455X (gfx1250) — matrix operand order repaired and checked by reading, not compiled, not run on hardware
//
#include <hip/hip_runtime.h>
#include <cstdint>

typedef __attribute__((ext_vector_type(16))) _Float16 v16h;
typedef __attribute__((ext_vector_type(8)))  float    v8f;

#define S_LEN 2048
#define D_DIM 128
#define BH    32
#define QT    64      // query rows per block (4 waves x 16)
#define NWAVE 4
#define QMAX4 7.0f    // 2^(4-1)-1
#define EPSQ  1e-8f
#define SNEG  (-1.0e30f)
#define LSTR  40      // LDS transpose row stride in halves (80B, conflict-free, 16B-aligned)

union FragH { v16h h; uint4 q[2]; };

__device__ inline float half_max16(float v) {
#pragma unroll
  for (int o = 1; o < 16; o <<= 1) v = fmaxf(v, __shfl_xor(v, o, 32));
  return v;
}
__device__ inline float full_max32(float v) {
#pragma unroll
  for (int o = 1; o < 32; o <<= 1) v = fmaxf(v, __shfl_xor(v, o, 32));
  return v;
}

// ---------- per-token 4-bit quant-dequant of K -> f16 row-major [BH,S,D]
__global__ __launch_bounds__(256) void prep_k_kernel(const float* __restrict__ K,
                                                     _Float16* __restrict__ Kdq) {
  const int wave = threadIdx.x >> 5, lane = threadIdx.x & 31;
  const size_t row = (size_t)blockIdx.x * 8 + wave;          // BH*S rows
  const float4 x = *(const float4*)(K + row * D_DIM + lane * 4);
  float am = fmaxf(fmaxf(fabsf(x.x), fabsf(x.y)), fmaxf(fabsf(x.z), fabsf(x.w)));
  am = full_max32(am);
  const float scale = fmaxf(am / QMAX4, EPSQ);
  const float inv = 1.0f / scale;
  union { _Float16 h[4]; uint2 u; } o;
  o.h[0] = (_Float16)(fminf(QMAX4, fmaxf(-QMAX4, rintf(x.x * inv))) * scale);
  o.h[1] = (_Float16)(fminf(QMAX4, fmaxf(-QMAX4, rintf(x.y * inv))) * scale);
  o.h[2] = (_Float16)(fminf(QMAX4, fmaxf(-QMAX4, rintf(x.z * inv))) * scale);
  o.h[3] = (_Float16)(fminf(QMAX4, fmaxf(-QMAX4, rintf(x.w * inv))) * scale);
  *(uint2*)(Kdq + row * D_DIM + lane * 4) = o.u;
}

// ---------- grouped (64) 4-bit quant-dequant of V -> f16 TRANSPOSED [BH,D,S]
__global__ __launch_bounds__(256) void prep_v_kernel(const float* __restrict__ V,
                                                     _Float16* __restrict__ VdqT) {
  const int wave = threadIdx.x >> 5, lane = threadIdx.x & 31;
  const size_t row = (size_t)blockIdx.x * 8 + wave;          // BH*S rows
  const int bh = (int)(row >> 11);
  const int s  = (int)(row & (S_LEN - 1));
  const float4 x = *(const float4*)(V + row * D_DIM + lane * 4);
  float am = fmaxf(fmaxf(fabsf(x.x), fabsf(x.y)), fmaxf(fabsf(x.z), fabsf(x.w)));
  am = half_max16(am);                                       // one 64-wide group per 16 lanes
  const float scale = fmaxf(am / QMAX4, EPSQ);
  const float inv = 1.0f / scale;
  float dq[4];
  dq[0] = fminf(QMAX4, fmaxf(-QMAX4, rintf(x.x * inv))) * scale;
  dq[1] = fminf(QMAX4, fmaxf(-QMAX4, rintf(x.y * inv))) * scale;
  dq[2] = fminf(QMAX4, fmaxf(-QMAX4, rintf(x.z * inv))) * scale;
  dq[3] = fminf(QMAX4, fmaxf(-QMAX4, rintf(x.w * inv))) * scale;
  _Float16* dst = VdqT + (size_t)bh * D_DIM * S_LEN + s;
  const int dbase = lane * 4;
#pragma unroll
  for (int i = 0; i < 4; ++i)
    dst[(size_t)(dbase + i) * S_LEN] = (_Float16)dq[i];
}

// ---------- fused attention: stats pass (WMMA) + recompute/normalize/PV pass (WMMA)
__global__ __launch_bounds__(128, 1)
void tq_attn_kernel(const float* __restrict__ Q,
                    const _Float16* __restrict__ Kdq,
                    const _Float16* __restrict__ VdqT,
                    float* __restrict__ Out,      // [BH,S,D]
                    float* __restrict__ W) {      // [BH,S,S]
  __shared__ __align__(16) _Float16 pbuf[NWAVE][16][LSTR];
  const int wave = threadIdx.x >> 5;
  const int lane = threadIdx.x & 31;
  const int h  = lane >> 4;       // K-half / row-half selector
  const int nn = lane & 15;       // A-row / B-col / C-col index
  const int bh = blockIdx.x / (S_LEN / QT);
  const int q0 = (blockIdx.x % (S_LEN / QT)) * QT + wave * 16;
  const int qmaxw = q0 + 15;
  const float scale = 0.08838834764831845f;      // 1/sqrt(128)
  const _Float16* Kbh = Kdq + (size_t)bh * S_LEN * D_DIM;
  float* Wbh = W + (size_t)bh * S_LEN * S_LEN;

  // ---- A fragments for Q (held for the whole kernel): 4 chunks of K=32
  const float* qrow = Q + ((size_t)bh * S_LEN + (q0 + nn)) * D_DIM;
  FragH afrag[4];
#pragma unroll
  for (int c = 0; c < 4; ++c) {
    const float* p0 = qrow + c * 32 + h * 8;
    const float* p1 = qrow + c * 32 + 16 + h * 8;
#pragma unroll
    for (int i = 0; i < 8; ++i) {
      afrag[c].h[i]     = (_Float16)p0[i];
      afrag[c].h[8 + i] = (_Float16)p1[i];
    }
  }

  // -------- pass A: stats only (no stores). Column-local online log-sum-exp.
  float m_[8], sum_[8];
#pragma unroll
  for (int v = 0; v < 8; ++v) { m_[v] = SNEG; sum_[v] = 0.f; }

  for (int j0 = 0; j0 <= qmaxw; j0 += 16) {
    const _Float16* krow = Kbh + (size_t)(j0 + nn) * D_DIM;
    __builtin_prefetch(krow + 16 * D_DIM, 0, 0);
    v8f c = {};
#pragma unroll
    for (int cc = 0; cc < 4; ++cc) {
      FragH b;
      b.q[0] = *(const uint4*)(krow + cc * 32 + h * 8);
      b.q[1] = *(const uint4*)(krow + cc * 32 + 16 + h * 8);
      c = __builtin_amdgcn_wmma_f32_16x16x32_f16(false, afrag[cc].h,
                                                 false, b.h,
                                                 (short)0, c, false, false);
    }
    const int col = j0 + nn;
#pragma unroll
    for (int v = 0; v < 8; ++v) {
      const bool ok = col <= (q0 + h * 8 + v);
      const float s = ok ? c[v] * scale : SNEG;
      const float nm = fmaxf(m_[v], s);
      const float p = ok ? __expf(s - nm) : 0.0f;
      sum_[v] = sum_[v] * __expf(m_[v] - nm) + p;
      m_[v] = nm;
    }
  }
  // merge (max,sum) pairs across the 16 lanes of each half (all finite -> no NaN)
#pragma unroll
  for (int o = 1; o < 16; o <<= 1) {
#pragma unroll
    for (int v = 0; v < 8; ++v) {
      const float om = __shfl_xor(m_[v], o, 32);
      const float os = __shfl_xor(sum_[v], o, 32);
      const float nm = fmaxf(m_[v], om);
      sum_[v] = sum_[v] * __expf(m_[v] - nm) + os * __expf(om - nm);
      m_[v] = nm;
    }
  }
  float inv_[8];
#pragma unroll
  for (int v = 0; v < 8; ++v) inv_[v] = 1.0f / sum_[v];

  // -------- pass B: recompute scores, normalize, write weights once, P.V
  v8f acc[8] = {};
  for (int j2 = 0; j2 <= qmaxw; j2 += 32) {
#pragma unroll
    for (int tt = 0; tt < 2; ++tt) {
      const int j0 = j2 + tt * 16;
      const _Float16* krow = Kbh + (size_t)(j0 + nn) * D_DIM;
      __builtin_prefetch(krow + 32 * D_DIM, 0, 0);
      v8f c = {};
#pragma unroll
      for (int cc = 0; cc < 4; ++cc) {
        FragH b;
        b.q[0] = *(const uint4*)(krow + cc * 32 + h * 8);
        b.q[1] = *(const uint4*)(krow + cc * 32 + 16 + h * 8);
        c = __builtin_amdgcn_wmma_f32_16x16x32_f16(false, afrag[cc].h,
                                                   false, b.h,
                                                   (short)0, c, false, false);
      }
      const int col = j0 + nn;
      float* wcol = Wbh + (size_t)(q0 + h * 8) * S_LEN + col;
#pragma unroll
      for (int v = 0; v < 8; ++v) {
        const bool ok = col <= (q0 + h * 8 + v);
        const float w = ok ? __expf(c[v] * scale - m_[v]) * inv_[v] : 0.0f;
        wcol[(size_t)v * S_LEN] = w;                       // final weights (single write)
        pbuf[wave][h * 8 + v][tt * 16 + nn] = (_Float16)w; // transpose staging
      }
    }
    // A fragment of P from LDS transpose buffer (lane row = nn)
    FragH a;
    a.q[0] = *(const uint4*)(&pbuf[wave][nn][h * 8]);
    a.q[1] = *(const uint4*)(&pbuf[wave][nn][16 + h * 8]);
#pragma unroll
    for (int t = 0; t < 8; ++t) {
      const _Float16* vrow = VdqT + ((size_t)bh * D_DIM + (t * 16 + nn)) * (size_t)S_LEN;
      FragH b;
      b.q[0] = *(const uint4*)(vrow + j2 + h * 8);
      b.q[1] = *(const uint4*)(vrow + j2 + 16 + h * 8);
      acc[t] = __builtin_amdgcn_wmma_f32_16x16x32_f16(false, a.h,
                                                      false, b.h,
                                                      (short)0, acc[t], false, false);
    }
  }

  // -------- zero-fill the untouched upper-triangle columns [cend, S)
  const int cend = ((qmaxw >> 5) + 1) << 5;
  const float4 z4 = make_float4(0.f, 0.f, 0.f, 0.f);
  for (int r = 0; r < 16; r += 2) {
    float* wr = Wbh + (size_t)(q0 + r + h) * S_LEN;
    for (int j = cend + nn * 4; j < S_LEN; j += 64)
      *(float4*)(wr + j) = z4;
  }

  // -------- write output [BH,S,D]
  float* orow = Out + ((size_t)bh * S_LEN + (q0 + h * 8)) * D_DIM + nn;
#pragma unroll
  for (int t = 0; t < 8; ++t)
#pragma unroll
    for (int v = 0; v < 8; ++v)
      orow[(size_t)v * D_DIM + t * 16] = acc[t][v];
}

extern "C" void kernel_launch(void* const* d_in, const int* in_sizes, int n_in,
                              void* d_out, int out_size, void* d_ws, size_t ws_size,
                              hipStream_t stream) {
  (void)in_sizes; (void)n_in; (void)out_size; (void)ws_size;
  const float* Q = (const float*)d_in[0];
  const float* K = (const float*)d_in[1];
  const float* V = (const float*)d_in[2];
  // d_in[3] = causal mask (implemented analytically)

  float* Out = (float*)d_out;                                      // [BH,S,D]
  float* W   = (float*)d_out + (size_t)BH * S_LEN * D_DIM;         // [BH,S,S]

  _Float16* Kdq  = (_Float16*)d_ws;                                // 16 MiB
  _Float16* VdqT = (_Float16*)((char*)d_ws +
                   (size_t)BH * S_LEN * D_DIM * sizeof(_Float16)); // 16 MiB

  const int rows = BH * S_LEN;                  // 65536 rows, 8 rows/block
  prep_k_kernel<<<dim3(rows / 8), dim3(256), 0, stream>>>(K, Kdq);
  prep_v_kernel<<<dim3(rows / 8), dim3(256), 0, stream>>>(V, VdqT);

  tq_attn_kernel<<<dim3(BH * (S_LEN / QT)), dim3(128), 0, stream>>>(Q, Kdq, VdqT, Out, W);
}
